// ScenarioDreamerEncoder_84439057039870
// MI455X (gfx1250) — matrix operand order repaired and checked by reading, not compiled, not run on hardware
//
#include <hip/hip_runtime.h>
#include <hip/hip_bf16.h>
#include <math.h>

typedef __bf16 bf16_t;
typedef bf16_t v16bf __attribute__((ext_vector_type(16)));
typedef bf16_t v8bf  __attribute__((ext_vector_type(8)));
typedef bf16_t v4bf  __attribute__((ext_vector_type(4)));
typedef float  v8f   __attribute__((ext_vector_type(8)));

// ---------------- model dims ----------------
enum {
    NL = 32768, NA = 16384, Bq = 256,
    HIDc = 256, AHIDc = 128, DFFc = 512, ADFFc = 256, Hc = 8,
    EA2A = 131072, EL2L = 262144, EL2A = 262144, EL2Q = 32768,
    LATc = 32, OUTD = 101
};

// =====================================================================
//  WMMA GEMM:  C[M,N] = opt_res + act( A[M,K] * W[K,N] + bias )
//  128x128 tile per 256-thread block (8 waves 4x2; each wave owns
//  32x64 = 8 accumulators -> 8 wmma per K-step). K stepped by 32.
//  Double-buffered LDS pipeline: one barrier per K-step, next tile's
//  global (or async global->LDS for bf16 A) loads overlap the WMMAs.
// =====================================================================
template <typename TA, typename TO>
__global__ __launch_bounds__(256) void gemm_wmma_kernel(
    const TA* __restrict__ A, const float* __restrict__ W,
    const float* __restrict__ bias, const float* __restrict__ res,
    TO* __restrict__ C, int M, int N, int K, int relu)
{
    __shared__ bf16_t As[2][128][32];    // A tiles (row stride 64B)
    __shared__ bf16_t BsT[2][128][32];   // W tiles transposed: BsT[n][k]

    const int tid  = threadIdx.x;
    const int lane = tid & 31;
    const int wv   = tid >> 5;
    const int wr   = wv >> 1;        // 0..3 -> 32-row slab
    const int wc   = wv & 1;         // 0..1 -> 64-col slab
    const int m0 = blockIdx.y * 128;
    const int n0 = blockIdx.x * 128;

    v8f acc[2][4];
#pragma unroll
    for (int i = 0; i < 2; ++i)
#pragma unroll
        for (int j = 0; j < 4; ++j)
            acc[i][j] = (v8f){0.f,0.f,0.f,0.f,0.f,0.f,0.f,0.f};

    // A fragment (16-bit 16x32 layout): lanes0-15 M=lane,K=0..7/16..23;
    // lanes16-31 M=lane-16,K=8..15/24..31
    const int mrow0 = wr * 32 + (lane & 15);
    const int mrow1 = mrow0 + 16;
    const int kbA   = (lane < 16) ? 0 : 8;
    // B fragment (32x16): lanes0-15 K=0..7/16..23, lanes16-31 K=8..15/24..31, N=lane%16
    const int kbB   = (lane < 16) ? 0 : 8;
    const int ncb   = wc * 64 + (lane & 15);

    const bool vecA = ((K & 3) == 0);
    const bool vecW = ((N & 3) == 0);
    const int nsteps = (K + 31) / 32;

    auto stageA = [&](int buf, int k0) {
        if constexpr (sizeof(TA) == 2) {
            if ((K & 7) == 0) {
                // direct bf16 copy: 512 16-byte chunks, async global->LDS
#pragma unroll
                for (int c = 0; c < 2; ++c) {
                    int idx = tid + c * 256;
                    int r   = idx >> 2;            // 4 chunks (32 elems) per row
                    int col = (idx & 3) * 8;
                    int gm  = m0 + r; if (gm >= M) gm = M - 1;
                    unsigned lds_off = (unsigned)(size_t)&As[buf][r][col];
                    unsigned voff    = (unsigned)(((size_t)gm * K + (k0 + col)) * 2u);
                    asm volatile("global_load_async_to_lds_b128 %0, %1, %2"
                                 :: "v"(lds_off), "v"(voff), "s"(A) : "memory");
                }
            } else {
                for (int i = tid; i < 128 * 32; i += 256) {
                    int r = i >> 5, c = i & 31;
                    int gm = m0 + r, gk = k0 + c;
                    float v = (gm < M && gk < K) ? (float)A[(size_t)gm * K + gk] : 0.0f;
                    As[buf][r][c] = (bf16_t)v;
                }
            }
        } else {
            if (vecA) {
#pragma unroll
                for (int c = 0; c < 4; ++c) {
                    int idx = tid + c * 256;
                    int r   = idx >> 3;            // 8 float4 per row
                    int col = (idx & 7) * 4;
                    int gm = m0 + r, gk = k0 + col;
                    float4 f = make_float4(0.f, 0.f, 0.f, 0.f);
                    if (gm < M && gk < K) f = *(const float4*)&A[(size_t)gm * K + gk];
                    v4bf o;
                    o[0] = (bf16_t)f.x; o[1] = (bf16_t)f.y;
                    o[2] = (bf16_t)f.z; o[3] = (bf16_t)f.w;
                    *(v4bf*)&As[buf][r][col] = o;
                }
            } else {
                for (int i = tid; i < 128 * 32; i += 256) {
                    int r = i >> 5, c = i & 31;
                    int gm = m0 + r, gk = k0 + c;
                    float v = (gm < M && gk < K) ? (float)A[(size_t)gm * K + gk] : 0.0f;
                    As[buf][r][c] = (bf16_t)v;
                }
            }
        }
    };

    auto stageW = [&](int buf, int k0) {
        if (vecW) {
#pragma unroll
            for (int c = 0; c < 4; ++c) {
                int idx = tid + c * 256;
                int r   = idx >> 5;                // k row; 32 float4 per row of 128
                int col = (idx & 31) * 4;
                int gk = k0 + r, gn = n0 + col;
                float4 f = make_float4(0.f, 0.f, 0.f, 0.f);
                if (gk < K && gn < N) f = *(const float4*)&W[(size_t)gk * N + gn];
                BsT[buf][col + 0][r] = (bf16_t)f.x;
                BsT[buf][col + 1][r] = (bf16_t)f.y;
                BsT[buf][col + 2][r] = (bf16_t)f.z;
                BsT[buf][col + 3][r] = (bf16_t)f.w;
            }
        } else {
            for (int i = tid; i < 32 * 128; i += 256) {
                int r = i >> 7, c = i & 127;       // r = k, c = n
                int gk = k0 + r, gn = n0 + c;
                float v = (gk < K && gn < N) ? W[(size_t)gk * N + gn] : 0.0f;
                BsT[buf][c][r] = (bf16_t)v;
            }
        }
    };

    stageA(0, 0);
    stageW(0, 0);

    for (int i = 0; i < nsteps; ++i) {
        const int buf = i & 1;
        if constexpr (sizeof(TA) == 2) {
            asm volatile("s_wait_asynccnt 0x0" ::: "memory");
        }
        __syncthreads();

        v8bf a0l = *(const v8bf*)&As[buf][mrow0][kbA];
        v8bf a0h = *(const v8bf*)&As[buf][mrow0][kbA + 16];
        v16bf a0 = __builtin_shufflevector(a0l, a0h, 0,1,2,3,4,5,6,7,8,9,10,11,12,13,14,15);
        v8bf a1l = *(const v8bf*)&As[buf][mrow1][kbA];
        v8bf a1h = *(const v8bf*)&As[buf][mrow1][kbA + 16];
        v16bf a1 = __builtin_shufflevector(a1l, a1h, 0,1,2,3,4,5,6,7,8,9,10,11,12,13,14,15);

        v16bf bf[4];
#pragma unroll
        for (int j = 0; j < 4; ++j) {
            const int n = ncb + j * 16;
            v8bf bl = *(const v8bf*)&BsT[buf][n][kbB];
            v8bf bh = *(const v8bf*)&BsT[buf][n][kbB + 16];
            bf[j] = __builtin_shufflevector(bl, bh, 0,1,2,3,4,5,6,7,8,9,10,11,12,13,14,15);
        }
#pragma unroll
        for (int j = 0; j < 4; ++j)
            acc[0][j] = __builtin_amdgcn_wmma_f32_16x16x32_bf16(false, a0, false, bf[j],
                                                                (short)0, acc[0][j], false, false);
#pragma unroll
        for (int j = 0; j < 4; ++j)
            acc[1][j] = __builtin_amdgcn_wmma_f32_16x16x32_bf16(false, a1, false, bf[j],
                                                                (short)0, acc[1][j], false, false);

        if (i + 1 < nsteps) {     // overlap next tile's loads with this tile's WMMAs
            stageA(buf ^ 1, (i + 1) * 32);
            stageW(buf ^ 1, (i + 1) * 32);
        }
    }

    // C/D layout: elem r -> M = r + (lane<16 ? 0 : 8), N = lane%16
    const int mb0 = wr * 32 + ((lane < 16) ? 0 : 8);
    const int nc  = lane & 15;
    auto emit = [&](const v8f& a, int mbase, int gn) {
        if (gn < N) {
#pragma unroll
            for (int r = 0; r < 8; ++r) {
                int gm = m0 + mbase + r;
                if (gm < M) {
                    float t = a[r];
                    if (bias) t += bias[gn];
                    if (relu) t = fmaxf(t, 0.0f);
                    if (res)  t += res[(size_t)gm * N + gn];
                    C[(size_t)gm * N + gn] = (TO)t;
                }
            }
        }
    };
#pragma unroll
    for (int j = 0; j < 4; ++j) {
        emit(acc[0][j], mb0,      n0 + wc * 64 + j * 16 + nc);
        emit(acc[1][j], mb0 + 16, n0 + wc * 64 + j * 16 + nc);
    }
}

// =====================================================================
//  Elementwise / reduction kernels
// =====================================================================
__global__ void fill_f32_kernel(float* p, float v, size_t n) {
    size_t i = (size_t)blockIdx.x * blockDim.x + threadIdx.x;
    if (i < n) p[i] = v;
}
__global__ void fill_u32_kernel(unsigned* p, unsigned v, size_t n) {
    size_t i = (size_t)blockIdx.x * blockDim.x + threadIdx.x;
    if (i < n) p[i] = v;
}
__global__ void tile_row_kernel(const float* __restrict__ row, float* __restrict__ out,
                                int R, int Dd) {
    int i = blockIdx.x * blockDim.x + threadIdx.x;
    if (i < R * Dd) out[i] = row[i % Dd];
}

__global__ void layernorm_kernel(const float* __restrict__ x, const float* __restrict__ g,
                                 const float* __restrict__ b, float* __restrict__ y,
                                 int N, int Dd) {
    int row = blockIdx.x * (blockDim.x >> 5) + (threadIdx.x >> 5);
    if (row >= N) return;
    int lane = threadIdx.x & 31;
    const float* xr = x + (size_t)row * Dd;
    float s = 0.f, s2 = 0.f;
    for (int j = lane; j < Dd; j += 32) { float v = xr[j]; s += v; s2 += v * v; }
    for (int o = 16; o; o >>= 1) { s += __shfl_xor(s, o, 32); s2 += __shfl_xor(s2, o, 32); }
    float mean = s / Dd;
    float var  = s2 / Dd - mean * mean;
    float rstd = rsqrtf(var + 1e-5f);
    float* yr = y + (size_t)row * Dd;
    for (int j = lane; j < Dd; j += 32) yr[j] = (xr[j] - mean) * rstd * g[j] + b[j];
}

// ordered-uint encoding so atomicMax(uint) == max over floats (incl. negatives)
__device__ __forceinline__ unsigned f32_ord(float f) {
    unsigned u = __float_as_uint(f);
    return (u & 0x80000000u) ? ~u : (u | 0x80000000u);
}
__device__ __forceinline__ float ord_f32(unsigned u) {
    return (u & 0x80000000u) ? __uint_as_float(u & 0x7fffffffu) : __uint_as_float(~u);
}

__global__ void edge_score_kernel(const float* __restrict__ q, const float* __restrict__ k,
                                  const bf16_t* __restrict__ ke,
                                  const int* __restrict__ src, const int* __restrict__ dst,
                                  int dst_off, float* __restrict__ score,
                                  unsigned* __restrict__ menc,
                                  int E, int Hh, int D, float scale) {
    int idx = blockIdx.x * blockDim.x + threadIdx.x;
    if (idx >= E * Hh) return;
    int e = idx / Hh, h = idx - e * Hh;
    int sr = src[e];
    int dr = dst[e] - dst_off;
    const float* qp = q + ((size_t)dr * Hh + h) * D;
    const float* kp = k + ((size_t)sr * Hh + h) * D;
    float s = 0.f;
    if (ke) {
        const bf16_t* kep = ke + ((size_t)e * Hh + h) * D;
        for (int d = 0; d < D; ++d) s += qp[d] * (kp[d] + (float)kep[d]);
    } else {
        for (int d = 0; d < D; ++d) s += qp[d] * kp[d];
    }
    s *= scale;
    score[idx] = s;
    atomicMax(&menc[(size_t)dr * Hh + h], f32_ord(s));
}

__global__ void finalize_m_kernel(const unsigned* __restrict__ menc, float* __restrict__ m,
                                  size_t n) {
    size_t i = (size_t)blockIdx.x * blockDim.x + threadIdx.x;
    if (i >= n) return;
    float f = ord_f32(menc[i]);
    m[i] = __builtin_isfinite(f) ? f : 0.0f;
}

__global__ void edge_expsum_kernel(float* __restrict__ score, const float* __restrict__ m,
                                   const int* __restrict__ dst, int dst_off,
                                   float* __restrict__ sbuf, int E, int Hh) {
    int idx = blockIdx.x * blockDim.x + threadIdx.x;
    if (idx >= E * Hh) return;
    int e = idx / Hh, h = idx - e * Hh;
    int dr = dst[e] - dst_off;
    float w = __expf(score[idx] - m[(size_t)dr * Hh + h]);
    score[idx] = w;
    atomicAdd(&sbuf[(size_t)dr * Hh + h], w);
}

__global__ void edge_accum_kernel(const float* __restrict__ score, const float* __restrict__ sbuf,
                                  const float* __restrict__ v, const bf16_t* __restrict__ ve,
                                  const int* __restrict__ src, const int* __restrict__ dst,
                                  int dst_off, float* __restrict__ msg, int E, int Hh, int D) {
    int idx = blockIdx.x * blockDim.x + threadIdx.x;
    if (idx >= E * Hh) return;
    int e = idx / Hh, h = idx - e * Hh;
    int sr = src[e];
    int dr = dst[e] - dst_off;
    float alpha = score[idx] / (sbuf[(size_t)dr * Hh + h] + 1e-9f);
    const float* vp = v + ((size_t)sr * Hh + h) * D;
    float* mp = msg + ((size_t)dr * Hh + h) * D;
    if (ve) {
        const bf16_t* vep = ve + ((size_t)e * Hh + h) * D;
        for (int d = 0; d < D; ++d) atomicAdd(&mp[d], alpha * (vp[d] + (float)vep[d]));
    } else {
        for (int d = 0; d < D; ++d) atomicAdd(&mp[d], alpha * vp[d]);
    }
}

__global__ void softmax_kernel(const float* __restrict__ in, float* __restrict__ out, int Dd) {
    int row = blockIdx.x;
    int lane = threadIdx.x;
    const float* x = in + (size_t)row * Dd;
    float* y = out + (size_t)row * Dd;
    float mx = -3.4e38f;
    for (int j = lane; j < Dd; j += 32) mx = fmaxf(mx, x[j]);
    for (int o = 16; o; o >>= 1) mx = fmaxf(mx, __shfl_xor(mx, o, 32));
    float s = 0.f;
    for (int j = lane; j < Dd; j += 32) s += __expf(x[j] - mx);
    for (int o = 16; o; o >>= 1) s += __shfl_xor(s, o, 32);
    float inv = 1.f / s;
    for (int j = lane; j < Dd; j += 32) y[j] = __expf(x[j] - mx) * inv;
}

// =====================================================================
//  Host-side orchestration
// =====================================================================
struct AttnP {
    const float *Wk_b, *Wk_w, *Wke_b, *Wke_w, *Wo_b, *Wo_w, *Wq_b, *Wq_w;
    const float *Wv_b, *Wv_w, *Wve_b, *Wve_w;
    const float *ffn1_b, *ffn1_w, *ffn2_b, *ffn2_w;
    const float *ln2_b, *ln2_g, *ln_dst_b, *ln_dst_g, *ln_src_b, *ln_src_g;
};

struct Scratch {
    float *hs, *hd, *q, *k, *v, *msg, *x, *h2, *ffnh, *score, *m, *s;
    unsigned* menc;
};

static inline void gemm_ff(hipStream_t st, const float* A, const float* W, const float* b,
                           const float* res, float* C, int M, int N, int K, bool relu) {
    dim3 g((N + 127) / 128, (M + 127) / 128);
    gemm_wmma_kernel<float, float><<<g, 256, 0, st>>>(A, W, b, res, C, M, N, K, relu ? 1 : 0);
}
static inline void gemm_fb(hipStream_t st, const float* A, const float* W, const float* b,
                           const float* res, bf16_t* C, int M, int N, int K, bool relu) {
    dim3 g((N + 127) / 128, (M + 127) / 128);
    gemm_wmma_kernel<float, bf16_t><<<g, 256, 0, st>>>(A, W, b, res, C, M, N, K, relu ? 1 : 0);
}
static inline void gemm_bb(hipStream_t st, const bf16_t* A, const float* W, const float* b,
                           const float* res, bf16_t* C, int M, int N, int K, bool relu) {
    dim3 g((N + 127) / 128, (M + 127) / 128);
    gemm_wmma_kernel<bf16_t, bf16_t><<<g, 256, 0, st>>>(A, W, b, res, C, M, N, K, relu ? 1 : 0);
}
static inline void ln_go(hipStream_t st, const float* x, const float* g, const float* b,
                         float* y, int N, int D) {
    layernorm_kernel<<<(N + 7) / 8, 256, 0, st>>>(x, g, b, y, N, D);
}
static inline void fill_f(hipStream_t st, float* p, float v, size_t n) {
    fill_f32_kernel<<<(unsigned)((n + 255) / 256), 256, 0, st>>>(p, v, n);
}
static inline void fill_u(hipStream_t st, unsigned* p, unsigned v, size_t n) {
    fill_u32_kernel<<<(unsigned)((n + 255) / 256), 256, 0, st>>>(p, v, n);
}

static void run_attention(hipStream_t st, const AttnP& p,
                          const float* xsrc, int n_src, int d_src,
                          const float* xdst_in, float* xdst_out, int n_dst, int d_dst,
                          const int* esrc, const int* edst, int E, int dst_off,
                          int heads, int d_ff, const bf16_t* ke, const bf16_t* ve,
                          const Scratch& S) {
    const int HD = d_dst, D = HD / heads;
    ln_go(st, xsrc, p.ln_src_g, p.ln_src_b, S.hs, n_src, d_src);
    ln_go(st, xdst_in, p.ln_dst_g, p.ln_dst_b, S.hd, n_dst, d_dst);
    gemm_ff(st, S.hd, p.Wq_w, p.Wq_b, nullptr, S.q, n_dst, HD, d_dst, false);
    gemm_ff(st, S.hs, p.Wk_w, p.Wk_b, nullptr, S.k, n_src, HD, d_src, false);
    gemm_ff(st, S.hs, p.Wv_w, p.Wv_b, nullptr, S.v, n_src, HD, d_src, false);
    fill_u(st, S.menc, 0u, (size_t)n_dst * heads);
    fill_f(st, S.s, 0.f, (size_t)n_dst * heads);
    fill_f(st, S.msg, 0.f, (size_t)n_dst * HD);
    int tot = E * heads;
    int blocks = (tot + 255) / 256;
    float scale = 1.0f / sqrtf((float)D);
    edge_score_kernel<<<blocks, 256, 0, st>>>(S.q, S.k, ke, esrc, edst, dst_off,
                                              S.score, S.menc, E, heads, D, scale);
    size_t nm = (size_t)n_dst * heads;
    finalize_m_kernel<<<(unsigned)((nm + 255) / 256), 256, 0, st>>>(S.menc, S.m, nm);
    edge_expsum_kernel<<<blocks, 256, 0, st>>>(S.score, S.m, edst, dst_off, S.s, E, heads);
    edge_accum_kernel<<<blocks, 256, 0, st>>>(S.score, S.s, S.v, ve, esrc, edst, dst_off,
                                              S.msg, E, heads, D);
    gemm_ff(st, S.msg, p.Wo_w, p.Wo_b, xdst_in, S.x, n_dst, d_dst, HD, false);
    ln_go(st, S.x, p.ln2_g, p.ln2_b, S.h2, n_dst, d_dst);
    gemm_ff(st, S.h2, p.ffn1_w, p.ffn1_b, nullptr, S.ffnh, n_dst, d_ff, d_dst, true);
    gemm_ff(st, S.ffnh, p.ffn2_w, p.ffn2_b, S.x, xdst_out, n_dst, d_dst, d_ff, false);
}

extern "C" void kernel_launch(void* const* d_in, const int* in_sizes, int n_in,
                              void* d_out, int out_size, void* d_ws, size_t ws_size,
                              hipStream_t stream) {
    (void)in_sizes; (void)out_size; (void)ws_size;
    if (n_in < 171) return;

    // ---------- unpack params (jax pytree order: sorted dict keys) ----------
    int cur = 0;
    auto nxt = [&]() -> const float* { return (const float*)d_in[cur++]; };
    auto unpack_attn = [&](bool edge) {
        AttnP p{};
        p.Wk_b = nxt(); p.Wk_w = nxt();
        if (edge) { p.Wke_b = nxt(); p.Wke_w = nxt(); }
        p.Wo_b = nxt(); p.Wo_w = nxt();
        p.Wq_b = nxt(); p.Wq_w = nxt();
        p.Wv_b = nxt(); p.Wv_w = nxt();
        if (edge) { p.Wve_b = nxt(); p.Wve_w = nxt(); }
        p.ffn1_b = nxt(); p.ffn1_w = nxt();
        p.ffn2_b = nxt(); p.ffn2_w = nxt();
        p.ln2_b = nxt(); p.ln2_g = nxt();
        p.ln_dst_b = nxt(); p.ln_dst_g = nxt();
        p.ln_src_b = nxt(); p.ln_src_g = nxt();
        return p;
    };

    const float* Qp     = nxt();                       // 'Q'
    const float* alv_b  = nxt(); const float* alv_w  = nxt();   // agent_log_var
    const float* am_bb  = nxt(); const float* am_bw  = nxt();   // agent_mlp.blocks[0]
    const float* am_fb  = nxt(); const float* am_fw  = nxt();   // agent_mlp.fc_in
    const float* amu_b  = nxt(); const float* amu_w  = nxt();   // agent_mu
    AttnP a2a_p[2], l2a_p[2], l2l_p[2];
    for (int bi = 0; bi < 2; ++bi) {                   // blocks: a2a, l2a, l2l (sorted)
        a2a_p[bi] = unpack_attn(false);
        l2a_p[bi] = unpack_attn(false);
        l2l_p[bi] = unpack_attn(true);
    }
    const float* cm_bb  = nxt(); const float* cm_bw  = nxt();   // conn_mlp.blocks[0]
    const float* cm_fb  = nxt(); const float* cm_fw  = nxt();   // conn_mlp.fc_in
    AttnP l2q_p = unpack_attn(false);                  // 'l2q'
    const float* llv_b  = nxt(); const float* llv_w  = nxt();   // lane_log_var
    const float* lm_bb  = nxt(); const float* lm_bw  = nxt();   // lane_mlp.blocks[0]
    const float* lm_fb  = nxt(); const float* lm_fw  = nxt();   // lane_mlp.fc_in
    const float* lmu_b  = nxt(); const float* lmu_w  = nxt();   // lane_mu
    const float* pr_b0b = nxt(); const float* pr_b0w = nxt();   // pred.blocks[0]
    const float* pr_b1b = nxt(); const float* pr_b1w = nxt();   // pred.blocks[1]
    const float* pr_fib = nxt(); const float* pr_fiw = nxt();   // pred.fc_in
    const float* pr_fob = nxt(); const float* pr_fow = nxt();   // pred.fc_out

    const float* x_agent = nxt();
    const float* x_lane  = nxt();
    const float* x_conn  = nxt();
    const int* a2a_e = (const int*)d_in[cur++];  // (2, EA2A) row-major
    const int* l2l_e = (const int*)d_in[cur++];
    const int* l2a_e = (const int*)d_in[cur++];
    const int* l2q_e = (const int*)d_in[cur++];
    // agent_batch (d_in[cur]) unused by reference

    // ---------- workspace carve ----------
    char* wbase = (char*)d_ws;
    size_t off = 0;
    auto carve = [&](size_t bytes) {
        void* p = wbase + off;
        off = (off + bytes + 255) & ~(size_t)255;
        return p;
    };
    float*  big0   = (float*)carve((size_t)EL2L * 256 * 4);   // conn f32 tmp; later ke/ve (bf16)
    bf16_t* connb  = (bf16_t*)carve((size_t)EL2L * 256 * 2);  // conn embedding (bf16)
    float*  laneA  = (float*)carve((size_t)NL * 256 * 4);
    float*  laneB  = (float*)carve((size_t)NL * 256 * 4);
    float*  agentA = (float*)carve((size_t)NA * 128 * 4);
    float*  agentB = (float*)carve((size_t)NA * 128 * 4);
    Scratch S;
    S.hs   = (float*)carve((size_t)NL * 256 * 4);
    S.hd   = (float*)carve((size_t)NL * 256 * 4);
    S.q    = (float*)carve((size_t)NL * 256 * 4);
    S.k    = (float*)carve((size_t)NL * 256 * 4);
    S.v    = (float*)carve((size_t)NL * 256 * 4);
    S.msg  = (float*)carve((size_t)NL * 256 * 4);
    S.x    = (float*)carve((size_t)NL * 256 * 4);
    S.h2   = (float*)carve((size_t)NL * 256 * 4);
    S.ffnh = (float*)carve((size_t)NL * 512 * 4);
    S.score = (float*)carve((size_t)EL2L * 8 * 4);
    S.m    = (float*)carve((size_t)NL * 8 * 4);
    S.s    = (float*)carve((size_t)NL * 8 * 4);
    S.menc = (unsigned*)carve((size_t)NL * 8 * 4);
    float* qembA = (float*)carve((size_t)Bq * 256 * 4);
    float* qembB = (float*)carve((size_t)Bq * 256 * 4);
    float* predA = (float*)carve((size_t)Bq * 256 * 4);
    float* predB = (float*)carve((size_t)Bq * 256 * 4);
    bf16_t* keb = (bf16_t*)big0;
    bf16_t* veb = (bf16_t*)((char*)big0 + (size_t)EL2L * 256 * 2);

    // ---------- output layout (return-tuple order, all f32) ----------
    float* out = (float*)d_out;
    float* out_agent_mu = out;
    float* out_lane_mu  = out_agent_mu + (size_t)NA * LATc;
    float* out_agent_lv = out_lane_mu + (size_t)NL * LATc;
    float* out_lane_lv  = out_agent_lv + (size_t)NA * LATc;
    float* out_logits   = out_lane_lv + (size_t)NL * LATc;
    float* out_prob     = out_logits + (size_t)Bq * OUTD;

    hipStream_t st = stream;

    // ---------- input embeddings (residual MLPs) ----------
    gemm_ff(st, x_agent, am_fw, am_fb, nullptr, agentA, NA, AHIDc, 10, true);
    gemm_ff(st, agentA, am_bw, am_bb, agentA, agentB, NA, AHIDc, AHIDc, true);
    float* agent = agentB; float* agentO = agentA;

    gemm_ff(st, x_lane, lm_fw, lm_fb, nullptr, laneA, NL, HIDc, 44, true);
    gemm_ff(st, laneA, lm_bw, lm_bb, laneA, laneB, NL, HIDc, HIDc, true);
    float* lane = laneB; float* laneO = laneA;

    gemm_ff(st, x_conn, cm_fw, cm_fb, nullptr, big0, EL2L, HIDc, 4, true);
    gemm_fb(st, big0, cm_bw, cm_bb, big0, connb, EL2L, HIDc, HIDc, true);  // -> bf16

    tile_row_kernel<<<(Bq * HIDc + 255) / 256, 256, 0, st>>>(Qp, qembA, Bq, HIDc);
    float* qemb = qembA; float* qembO = qembB;

    // ---------- transformer blocks ----------
    for (int bi = 0; bi < 2; ++bi) {
        // l2l (edge_feat = conn): per-edge key/value contributions in bf16
        gemm_bb(st, connb, l2l_p[bi].Wke_w, l2l_p[bi].Wke_b, nullptr, keb, EL2L, HIDc, HIDc, false);
        gemm_bb(st, connb, l2l_p[bi].Wve_w, l2l_p[bi].Wve_b, nullptr, veb, EL2L, HIDc, HIDc, false);
        run_attention(st, l2l_p[bi], lane, NL, HIDc, lane, laneO, NL, HIDc,
                      l2l_e, l2l_e + EL2L, EL2L, 0, Hc, DFFc, keb, veb, S);
        { float* t = lane; lane = laneO; laneO = t; }

        run_attention(st, a2a_p[bi], agent, NA, AHIDc, agent, agentO, NA, AHIDc,
                      a2a_e, a2a_e + EA2A, EA2A, 0, Hc, ADFFc, nullptr, nullptr, S);
        { float* t = agent; agent = agentO; agentO = t; }

        run_attention(st, l2a_p[bi], lane, NL, HIDc, agent, agentO, NA, AHIDc,
                      l2a_e, l2a_e + EL2A, EL2A, 0, Hc, ADFFc, nullptr, nullptr, S);
        { float* t = agent; agent = agentO; agentO = t; }

        // l2q over concat(lane, q_emb): src rows are lanes, dst rows are queries (offset NL)
        run_attention(st, l2q_p, lane, NL, HIDc, qemb, qembO, Bq, HIDc,
                      l2q_e, l2q_e + EL2Q, EL2Q, NL, Hc, DFFc, nullptr, nullptr, S);
        { float* t = qemb; qemb = qembO; qembO = t; }
    }

    // ---------- prediction head ----------
    gemm_ff(st, qemb, pr_fiw, pr_fib, nullptr, predA, Bq, HIDc, HIDc, true);
    gemm_ff(st, predA, pr_b0w, pr_b0b, predA, predB, Bq, HIDc, HIDc, true);
    gemm_ff(st, predB, pr_b1w, pr_b1b, predB, predA, Bq, HIDc, HIDc, true);
    gemm_ff(st, predA, pr_fow, pr_fob, nullptr, out_logits, Bq, OUTD, HIDc, false);
    softmax_kernel<<<Bq, 32, 0, st>>>(out_logits, out_prob, OUTD);

    // ---------- latent heads ----------
    gemm_ff(st, agent, amu_w, amu_b, nullptr, out_agent_mu, NA, LATc, AHIDc, false);
    gemm_ff(st, lane,  lmu_w, lmu_b, nullptr, out_lane_mu,  NL, LATc, HIDc, false);
    gemm_ff(st, agent, alv_w, alv_b, nullptr, out_agent_lv, NA, LATc, AHIDc, false);
    gemm_ff(st, lane,  llv_w, llv_b, nullptr, out_lane_lv,  NL, LATc, HIDc, false);
}
